// DualEdgeEGNN_73641509257760
// MI455X (gfx1250) — matrix operand order repaired and checked by reading, not compiled, not run on hardware
//
#include <hip/hip_runtime.h>
#include <hip/hip_bf16.h>
#include <math.h>

typedef _Float16 half_t;
typedef __attribute__((ext_vector_type(16))) _Float16 v16h;
typedef __attribute__((ext_vector_type(8)))  float    v8f;

union FragA { v16h h; unsigned int u[8]; };

// ---- CDNA5 async global->LDS DMA (ASYNCcnt-tracked), per ISA 08_async_tensor §4.
// Generic shared-aperture addresses carry the LDS byte offset in addr[31:0].
__device__ __forceinline__ void async_copy_b128(unsigned lds_off, const void* gptr) {
  asm volatile("global_load_async_to_lds_b128 %0, %1, off"
               :: "v"(lds_off), "v"((unsigned long long)(uintptr_t)gptr)
               : "memory");
}
__device__ __forceinline__ void wait_async0() {
  asm volatile("s_wait_asynccnt 0" ::: "memory");
}

// ---------------------------------------------------------------------------
// Weight packing: f32 row-major (Kreal x Nreal) -> f16 WMMA B-fragment tiles.
// Tile (kt,nt): lane l holds column n = nt*16 + (l&15); its 16 halves are
// K = kt*32 + e + 16*(l>>4) for e = 0..15 (contiguous k order, same as the A fragment loads).
// Linear offset = ((kt*Ntiles + nt)*32 + lane)*16 + e.
// ---------------------------------------------------------------------------
__global__ void pack_weight_kernel(const float* __restrict__ W, half_t* __restrict__ Bp,
                                   int Kreal, int Nreal, int Kpad, int Npad) {
  int idx = blockIdx.x * 256 + threadIdx.x;
  int total = Kpad * Npad;
  if (idx >= total) return;
  int e      = idx & 15;
  int lane   = (idx >> 4) & 31;
  int tile   = idx >> 9;
  int Ntiles = Npad >> 4;
  int kt = tile / Ntiles;
  int nt = tile - kt * Ntiles;
  int n = nt * 16 + (lane & 15);
  int k = kt * 32 + e + ((lane >> 4) << 4);
  float v = (k < Kreal && n < Nreal) ? W[(size_t)k * Nreal + n] : 0.f;
  Bp[idx] = (half_t)v;
}

// ---------------------------------------------------------------------------
// Tiled WMMA GEMM:  C[M x Npad] = act( A[M x Kpad](f16) @ Bpacked + bias )
// 256 thr = 8 waves. Block tile 128(M) x 64(N); wave tile 16 x 64.
// Double-buffered LDS staging of A (128x32) and B (4 tiles, 4KB) via
// global_load_async_to_lds_b128; the 4 back-to-back WMMAs overlap next DMA.
// ACT: 0=none 1=silu 2=relu.  W16: write f16.  W32: 0=no,1=write,2=accum f32.
// M%128==0, Kpad%32==0, Npad%64==0 (guaranteed by caller padding).
// ---------------------------------------------------------------------------
template<int ACT, int W16, int W32>
__global__ __launch_bounds__(256)
void gemm_kernel(const half_t* __restrict__ A, const half_t* __restrict__ Bp,
                 const float* __restrict__ bias,
                 half_t* __restrict__ C16, float* __restrict__ C32,
                 int M, int Kpad, int Nreal, int Npad) {
  __shared__ __align__(16) half_t sA[2][128 * 32];
  __shared__ __align__(16) half_t sB[2][2048];
  const int tid  = threadIdx.x;
  const int wave = tid >> 5;
  const int lane = tid & 31;
  const int mBase = blockIdx.x * 128;
  const int nBase = blockIdx.y * 64;
  const int Ntiles = Npad >> 4;
  const int r = lane & 15;
  const int khalf = (lane >> 4) << 4;
  const int KT = Kpad >> 5;

  v8f acc[4] = {};

  // stage A tile (512 x 16B slots, 2/thread) + B slice (256 x 16B, 1/thread)
  auto stage = [&](int buf, int kt) {
    unsigned aBase = (unsigned)(uintptr_t)&sA[buf][0];
    int i0 = tid * 2;
#pragma unroll
    for (int s = i0; s < i0 + 2; ++s) {
      int row = s >> 2, q = s & 3;
      async_copy_b128(aBase + s * 16,
                      A + (size_t)(mBase + row) * Kpad + kt * 32 + q * 8);
    }
    unsigned bBase = (unsigned)(uintptr_t)&sB[buf][0];
    async_copy_b128(bBase + tid * 16,
                    Bp + ((size_t)kt * Ntiles + (nBase >> 4)) * 512 + tid * 8);
  };

  stage(0, 0);

  for (int kt = 0; kt < KT; ++kt) {
    wait_async0();       // this wave's staging of buf kt&1 done
    __syncthreads();     // everyone's staging done; prior compute done
    if (kt + 1 < KT) stage((kt + 1) & 1, kt + 1);  // DMA next while computing
    const int buf = kt & 1;

    // issue all fragment loads first, then run WMMAs back-to-back
    FragA fa;
#pragma unroll
    for (int j = 0; j < 8; ++j) {
      int k = 2 * j + khalf;
      fa.u[j] = *(const unsigned int*)(&sA[buf][(wave * 16 + r) * 32 + k]);
    }
    v16h fb[4];
#pragma unroll
    for (int t = 0; t < 4; ++t)
      fb[t] = *(const v16h*)(&sB[buf][(t * 32 + lane) * 16]);
#pragma unroll
    for (int t = 0; t < 4; ++t)
      acc[t] = __builtin_amdgcn_wmma_f32_16x16x32_f16(
          false, fa.h, false, fb[t], (short)0, acc[t], false, false);
  }

  // store: acc[t][rr] -> (m = mBase+wave*16+rr+(lane<16?0:8), n = nBase+t*16+(lane&15))
  const int nLane = lane & 15;
  const int mOff = (lane < 16) ? 0 : 8;
#pragma unroll
  for (int t = 0; t < 4; ++t) {
    int n = nBase + t * 16 + nLane;
    float bv = (n < Nreal && bias) ? bias[n] : 0.f;
#pragma unroll
    for (int rr = 0; rr < 8; ++rr) {
      int m = mBase + wave * 16 + rr + mOff;
      float v = acc[t][rr] + bv;
      if (ACT == 1) v = v / (1.f + __expf(-v));
      else if (ACT == 2) v = fmaxf(v, 0.f);
      if (n >= Nreal) v = 0.f;
      size_t o = (size_t)m * Npad + n;
      if (W16) C16[o] = (half_t)v;
      if (W32 == 1) C32[o] = v;
      else if (W32 == 2) C32[o] += v;
    }
  }
}

// ---------------------------------------------------------------------------
// Timestep embedding + 3-layer MLP. One block (128 thr) per graph.
// ---------------------------------------------------------------------------
__global__ void temb_kernel(const int* __restrict__ time_step,
                            const float* __restrict__ d0b, const float* __restrict__ d0w,
                            const float* __restrict__ d1b, const float* __restrict__ d1w,
                            const float* __restrict__ pjb, const float* __restrict__ pjw,
                            float* __restrict__ temb) {
  __shared__ float se[128];
  __shared__ float sh[512];
  __shared__ float sh2[512];
  int g = blockIdx.x, t = threadIdx.x;
  float ts = (float)time_step[g];
  {
    const float negln = -9.210340371976184f / 63.f; // -log(10000)/(half-1)
    if (t < 64) { float f = __expf(negln * t);        se[t] = __sinf(ts * f); }
    else        { float f = __expf(negln * (t - 64)); se[t] = __cosf(ts * f); }
  }
  __syncthreads();
#pragma unroll
  for (int rT = 0; rT < 4; ++rT) {
    int o = t + 128 * rT;
    float s = d0b[o];
    for (int k = 0; k < 128; ++k) s += se[k] * d0w[k * 512 + o];
    sh[o] = fmaxf(s, 0.f);
  }
  __syncthreads();
#pragma unroll
  for (int rT = 0; rT < 4; ++rT) {
    int o = t + 128 * rT;
    float s = d1b[o];
    for (int k = 0; k < 512; ++k) s += sh[k] * d1w[k * 512 + o];
    sh2[o] = fmaxf(s, 0.f);
  }
  __syncthreads();
  {
    float s = pjb[t];
    for (int k = 0; k < 512; ++k) s += sh2[k] * pjw[k * 128 + t];
    temb[g * 128 + t] = s;
  }
}

// ---------------------------------------------------------------------------
// edge_attr = mlp1(relu(mlp0(len))) * bond_emb[etype] + temb[graph]  (f16 out)
// ---------------------------------------------------------------------------
__global__ void edge_attr_kernel(const float* __restrict__ edge_length,
                                 const int* __restrict__ edge_type,
                                 const int* __restrict__ ei_row,
                                 const int* __restrict__ batch,
                                 const float* __restrict__ temb,
                                 const float* __restrict__ m0b, const float* __restrict__ m0w,
                                 const float* __restrict__ m1b, const float* __restrict__ m1w,
                                 const float* __restrict__ bond_emb,
                                 half_t* __restrict__ attr16) {
  __shared__ float sd[128];
  int e = blockIdx.x, t = threadIdx.x;
  float len = edge_length[e];
  sd[t] = fmaxf(len * m0w[t] + m0b[t], 0.f);
  __syncthreads();
  float s = m1b[t];
  for (int k = 0; k < 128; ++k) s += sd[k] * m1w[k * 128 + t];
  int bt = edge_type[e];
  int g = batch[ei_row[e]];
  float a = s * bond_emb[(size_t)bt * 128 + t] + temb[g * 128 + t];
  attr16[(size_t)e * 128 + t] = (half_t)a;
}

// x = concat(atom_type[16], temb[batch][128]) padded to 160, f16
__global__ void build_x_kernel(const float* __restrict__ atom_type,
                               const float* __restrict__ temb,
                               const int* __restrict__ batch,
                               half_t* __restrict__ x16, int N) {
  int idx = blockIdx.x * 256 + threadIdx.x;
  if (idx >= N * 160) return;
  int n = idx / 160, c = idx - n * 160;
  float v = 0.f;
  if (c < 16) v = atom_type[n * 16 + c];
  else if (c < 144) v = temb[batch[n] * 128 + (c - 16)];
  x16[idx] = (half_t)v;
}

// m_in = [feats_i(128) | feats_j(128) | rel_dist(1) | edge_attr(128) | pad->416]
__global__ void build_min_kernel(const half_t* __restrict__ feats16,
                                 const half_t* __restrict__ attr16,
                                 const float* __restrict__ coors,
                                 const int* __restrict__ ei_row, const int* __restrict__ ei_col,
                                 half_t* __restrict__ m_in) {
  int e = blockIdx.x, t = threadIdx.x; // 128 threads
  int r = ei_row[e], c = ei_col[e];
  half_t* o = m_in + (size_t)e * 416;
  o[t]       = feats16[(size_t)r * 128 + t];
  o[128 + t] = feats16[(size_t)c * 128 + t];
  o[257 + t] = attr16[(size_t)e * 128 + t];
  if (t == 0) {
    float dx = coors[r * 3 + 0] - coors[c * 3 + 0];
    float dy = coors[r * 3 + 1] - coors[c * 3 + 1];
    float dz = coors[r * 3 + 2] - coors[c * 3 + 2];
    o[256] = (half_t)(dx * dx + dy * dy + dz * dz);
  }
  if (t < 31) o[385 + t] = (half_t)0.f;
}

// coor_w = c1(ch), gate = sigmoid(gate_w . m). One wave per edge.
__global__ void edge_scalar_kernel(const half_t* __restrict__ m16,
                                   const half_t* __restrict__ ch,
                                   const float* __restrict__ wc1, const float* __restrict__ bc1,
                                   const float* __restrict__ wg,  const float* __restrict__ bg,
                                   float* __restrict__ coorw, float* __restrict__ gate, int E) {
  int e = blockIdx.x * 8 + (threadIdx.x >> 5);
  int lane = threadIdx.x & 31;
  if (e >= E) return;
  float s1 = 0.f, s2 = 0.f;
  for (int k = lane; k < 512; k += 32) s1 += (float)ch[(size_t)e * 512 + k] * wc1[k];
  for (int k = lane; k < 128; k += 32) s2 += (float)m16[(size_t)e * 128 + k] * wg[k];
#pragma unroll
  for (int off = 16; off; off >>= 1) {
    s1 += __shfl_xor(s1, off, 32);
    s2 += __shfl_xor(s2, off, 32);
  }
  if (lane == 0) {
    coorw[e] = s1 + bc1[0];
    gate[e]  = 1.f / (1.f + __expf(-(s2 + bg[0])));
  }
}

// Per-node aggregation (degree is exactly 16, edges contiguous per node):
// m_i = sum_e m*gate ; coors_out = coors + sum_e coorw*scale*rel/max(|rel|,1e-8) / 16
__global__ void aggregate_kernel(const half_t* __restrict__ m16,
                                 const float* __restrict__ gate,
                                 const float* __restrict__ coorw,
                                 const float* __restrict__ coors,
                                 float* __restrict__ coors_out,
                                 float* __restrict__ m_i,
                                 const int* __restrict__ ei_col,
                                 const float* __restrict__ scale_ptr) {
  int node = blockIdx.x, t = threadIdx.x; // 128 threads
  int ebase = node * 16;
  float acc = 0.f;
  for (int j = 0; j < 16; ++j) {
    int e = ebase + j;
    acc += (float)m16[(size_t)e * 128 + t] * gate[e];
  }
  m_i[(size_t)node * 128 + t] = acc;
  if (t == 0) {
    float scale = scale_ptr[0];
    float px = coors[node * 3 + 0], py = coors[node * 3 + 1], pz = coors[node * 3 + 2];
    float cx = 0.f, cy = 0.f, cz = 0.f;
    for (int j = 0; j < 16; ++j) {
      int e = ebase + j;
      int cc = ei_col[e];
      float dx = px - coors[cc * 3 + 0];
      float dy = py - coors[cc * 3 + 1];
      float dz = pz - coors[cc * 3 + 2];
      float nrm = fmaxf(sqrtf(dx * dx + dy * dy + dz * dz), 1e-8f);
      float w = coorw[e] * scale / nrm;
      cx += w * dx; cy += w * dy; cz += w * dz;
    }
    coors_out[node * 3 + 0] = px + cx * (1.f / 16.f);
    coors_out[node * 3 + 1] = py + cy * (1.f / 16.f);
    coors_out[node * 3 + 2] = pz + cz * (1.f / 16.f);
  }
}

// nin = concat(feats16, (f16)m_i)  [N x 256]
__global__ void build_nin_kernel(const half_t* __restrict__ feats16,
                                 const float* __restrict__ m_i,
                                 half_t* __restrict__ nin, int N) {
  int idx = blockIdx.x * 256 + threadIdx.x;
  if (idx >= N * 256) return;
  int n = idx >> 8, c = idx & 255;
  nin[idx] = (c < 128) ? feats16[(size_t)n * 128 + c]
                       : (half_t)m_i[(size_t)n * 128 + (c - 128)];
}

__global__ void refresh_f16_kernel(const float* __restrict__ src, half_t* __restrict__ dst, int total) {
  int idx = blockIdx.x * 256 + threadIdx.x;
  if (idx < total) dst[idx] = (half_t)src[idx];
}

// pair = [feats_i * feats_j (128) | edge_attr (128)]
__global__ void build_pair_kernel(const half_t* __restrict__ feats16,
                                  const half_t* __restrict__ attr16,
                                  const int* __restrict__ ei_row, const int* __restrict__ ei_col,
                                  half_t* __restrict__ pair, int E) {
  int idx = blockIdx.x * 256 + threadIdx.x;
  if (idx >= E * 256) return;
  int e = idx >> 8, c = idx & 255;
  if (c < 128) {
    int r = ei_row[e], cl = ei_col[e];
    float v = (float)feats16[(size_t)r * 128 + c] * (float)feats16[(size_t)cl * 128 + c];
    pair[idx] = (half_t)v;
  } else {
    pair[idx] = attr16[(size_t)e * 128 + (c - 128)];
  }
}

// final: out[e] = y2[e] . w + b  (one wave per edge)
__global__ void final_kernel(const half_t* __restrict__ y2,
                             const float* __restrict__ w, const float* __restrict__ b,
                             float* __restrict__ out, int E) {
  int e = blockIdx.x * 8 + (threadIdx.x >> 5);
  int lane = threadIdx.x & 31;
  if (e >= E) return;
  float s = 0.f;
  for (int k = lane; k < 64; k += 32) s += (float)y2[(size_t)e * 64 + k] * w[k];
#pragma unroll
  for (int off = 16; off; off >>= 1) s += __shfl_xor(s, off, 32);
  if (lane == 0) out[e] = s + b[0];
}

// pass-through tuple tail: edge_index, edge_type, edge_length after edge_inv
__global__ void tail_kernel(const int* __restrict__ eidx, const int* __restrict__ etype,
                            const float* __restrict__ elen, float* __restrict__ out,
                            int E, int out_size) {
  int i = blockIdx.x * 256 + threadIdx.x;
  if (i < 2 * E && (E + i) < out_size)       out[E + i]     = (float)eidx[i];
  if (i < E && (3 * E + i) < out_size)       out[3 * E + i] = (float)etype[i];
  if (i < E && (4 * E + i) < out_size)       out[4 * E + i] = elen[i];
}

// ---------------------------------------------------------------------------
// Host orchestration
// ---------------------------------------------------------------------------
extern "C" void kernel_launch(void* const* d_in, const int* in_sizes, int n_in,
                              void* d_out, int out_size, void* d_ws, size_t ws_size,
                              hipStream_t stream) {
  (void)n_in; (void)ws_size;
  const int E = in_sizes[8];        // edge_length: [E,1]
  const int N = in_sizes[0] / 16;   // atom_type: [N,16]
  const int G = in_sizes[5];        // time_step: [G]

  // tensor inputs
  const float* atom_type = (const float*)d_in[0];
  const float* pos       = (const float*)d_in[1];
  const int*   batch     = (const int*)d_in[4];
  const int*   tstep     = (const int*)d_in[5];
  const int*   etype     = (const int*)d_in[6];
  const int*   eidx      = (const int*)d_in[7];   // [2,E]
  const float* elen      = (const float*)d_in[8];
  const int*   ei_row = eidx;
  const int*   ei_col = eidx + E;

  // params (pytree, dict keys sorted)
  const float* bond_emb = (const float*)d_in[9];
  const float* d0b = (const float*)d_in[10]; const float* d0w = (const float*)d_in[11];
  const float* d1b = (const float*)d_in[12]; const float* d1w = (const float*)d_in[13];
  const float* d2b = (const float*)d_in[14]; const float* d2w = (const float*)d_in[15];
  const float* em0b = (const float*)d_in[16]; const float* em0w = (const float*)d_in[17];
  const float* em1b = (const float*)d_in[18]; const float* em1w = (const float*)d_in[19];
  const float* embb = (const float*)d_in[20]; const float* embw = (const float*)d_in[21];
  // layer L leaf k (0..14): c0b,c0w,c1b,c1w,coors_scale,e0b,e0w,e1b,e1w,gb,gw,n0b,n0w,n1b,n1w
#define LPAR(L, k) ((const float*)d_in[22 + 15 * (L) + (k)])
  const float* td0b = (const float*)d_in[112]; const float* td0w = (const float*)d_in[113];
  const float* td1b = (const float*)d_in[114]; const float* td1w = (const float*)d_in[115];
  const float* tpb  = (const float*)d_in[116]; const float* tpw  = (const float*)d_in[117];

  // padded GEMM dims
  const int K1 = 416, N1 = 832;   // e0: 385 -> 770
  const int K2 = 832, N2 = 128;   // e1

  // workspace carve-up
  char* ws = (char*)d_ws;
  size_t off = 0;
  auto carve = [&](size_t bytes) -> char* {
    char* p = ws + off;
    off += (bytes + 255) & ~(size_t)255;
    return p;
  };
  half_t* W_E0  = (half_t*)carve((size_t)6 * K1 * N1 * 2);
  half_t* W_E1  = (half_t*)carve((size_t)6 * K2 * N2 * 2);
  half_t* W_C0  = (half_t*)carve((size_t)6 * 128 * 512 * 2);
  half_t* W_N0  = (half_t*)carve((size_t)6 * 256 * 256 * 2);
  half_t* W_N1  = (half_t*)carve((size_t)6 * 256 * 128 * 2);
  half_t* W_EMB = (half_t*)carve((size_t)160 * 128 * 2);
  half_t* W_D0  = (half_t*)carve((size_t)256 * 128 * 2);
  half_t* W_D1  = (half_t*)carve((size_t)128 * 64 * 2);
  float*  temb    = (float*)carve((size_t)G * 128 * 4);
  half_t* attr16  = (half_t*)carve((size_t)E * 128 * 2);
  float*  feats32 = (float*)carve((size_t)N * 128 * 4);
  half_t* feats16 = (half_t*)carve((size_t)N * 128 * 2);
  float*  coorsA  = (float*)carve((size_t)N * 3 * 4);
  float*  coorsB  = (float*)carve((size_t)N * 3 * 4);
  half_t* x16     = (half_t*)carve((size_t)N * 160 * 2);
  half_t* m_in    = (half_t*)carve((size_t)E * 416 * 2);   // reused as pair16 (E*256)
  half_t* h1      = (half_t*)carve((size_t)E * 832 * 2);   // reused: ch(E*512), y1, y2
  half_t* m16     = (half_t*)carve((size_t)E * 128 * 2);
  float*  coorw   = (float*)carve((size_t)E * 4);
  float*  gate    = (float*)carve((size_t)E * 4);
  float*  m_i     = (float*)carve((size_t)N * 128 * 4);
  half_t* nin16   = (half_t*)carve((size_t)N * 256 * 2);
  half_t* nh16    = (half_t*)carve((size_t)N * 256 * 2);
  half_t* ch16   = h1;                                  // c0 output E x 512
  half_t* pair16 = m_in;                                // E x 256
  half_t* y1     = h1;                                  // E x 128
  half_t* y2     = (half_t*)((char*)h1 + ((size_t)E * 256 * 2)); // E x 64

  auto pack = [&](const float* W, half_t* dst, int Kr, int Nr, int Kp, int Np) {
    int total = Kp * Np;
    pack_weight_kernel<<<(total + 255) / 256, 256, 0, stream>>>(W, dst, Kr, Nr, Kp, Np);
  };

  // ---- pack all weights ----
  pack(embw, W_EMB, 144, 128, 160, 128);
  pack(d0w, W_D0, 256, 128, 256, 128);
  pack(d1w, W_D1, 128, 64, 128, 64);
  for (int L = 0; L < 6; ++L) {
    pack(LPAR(L, 6),  W_E0 + (size_t)L * K1 * N1, 385, 770, K1, N1);
    pack(LPAR(L, 8),  W_E1 + (size_t)L * K2 * N2, 770, 128, K2, N2);
    pack(LPAR(L, 1),  W_C0 + (size_t)L * 128 * 512, 128, 512, 128, 512);
    pack(LPAR(L, 12), W_N0 + (size_t)L * 256 * 256, 256, 256, 256, 256);
    pack(LPAR(L, 14), W_N1 + (size_t)L * 256 * 128, 256, 128, 256, 128);
  }

  // ---- timestep embedding ----
  temb_kernel<<<G, 128, 0, stream>>>(tstep, td0b, td0w, td1b, td1w, tpb, tpw, temb);

  // ---- edge encoder ----
  edge_attr_kernel<<<E, 128, 0, stream>>>(elen, etype, ei_row, batch, temb,
                                          em0b, em0w, em1b, em1w, bond_emb, attr16);

  // ---- embed: feats = x @ W_emb + b ----
  build_x_kernel<<<(N * 160 + 255) / 256, 256, 0, stream>>>(atom_type, temb, batch, x16, N);
  gemm_kernel<0, 1, 1><<<dim3(N / 128, 128 / 64), 256, 0, stream>>>(
      x16, W_EMB, embb, feats16, feats32, N, 160, 128, 128);

  // ---- EGNN layers ----
  const float* coors_cur = pos;
  float* coors_nxt = coorsA;
  for (int L = 0; L < 6; ++L) {
    build_min_kernel<<<E, 128, 0, stream>>>(feats16, attr16, coors_cur, ei_row, ei_col, m_in);
    gemm_kernel<1, 1, 0><<<dim3(E / 128, N1 / 64), 256, 0, stream>>>(
        m_in, W_E0 + (size_t)L * K1 * N1, LPAR(L, 5), h1, nullptr, E, K1, 770, N1);
    gemm_kernel<1, 1, 0><<<dim3(E / 128, N2 / 64), 256, 0, stream>>>(
        h1, W_E1 + (size_t)L * K2 * N2, LPAR(L, 7), m16, nullptr, E, K2, 128, N2);
    gemm_kernel<1, 1, 0><<<dim3(E / 128, 512 / 64), 256, 0, stream>>>(
        m16, W_C0 + (size_t)L * 128 * 512, LPAR(L, 0), ch16, nullptr, E, 128, 512, 512);
    edge_scalar_kernel<<<(E + 7) / 8, 256, 0, stream>>>(
        m16, ch16, LPAR(L, 3), LPAR(L, 2), LPAR(L, 10), LPAR(L, 9), coorw, gate, E);
    aggregate_kernel<<<N, 128, 0, stream>>>(
        m16, gate, coorw, coors_cur, coors_nxt, m_i, ei_col, LPAR(L, 4));
    build_nin_kernel<<<(N * 256 + 255) / 256, 256, 0, stream>>>(feats16, m_i, nin16, N);
    gemm_kernel<1, 1, 0><<<dim3(N / 128, 256 / 64), 256, 0, stream>>>(
        nin16, W_N0 + (size_t)L * 256 * 256, LPAR(L, 11), nh16, nullptr, N, 256, 256, 256);
    gemm_kernel<0, 0, 2><<<dim3(N / 128, 128 / 64), 256, 0, stream>>>(
        nh16, W_N1 + (size_t)L * 256 * 128, LPAR(L, 13), nullptr, feats32, N, 256, 128, 128);
    refresh_f16_kernel<<<(N * 128 + 255) / 256, 256, 0, stream>>>(feats32, feats16, N * 128);
    // ping-pong coordinates
    coors_cur = coors_nxt;
    coors_nxt = (coors_nxt == coorsA) ? coorsB : coorsA;
  }

  // ---- final distance MLP ----
  build_pair_kernel<<<(E * 256 + 255) / 256, 256, 0, stream>>>(
      feats16, attr16, ei_row, ei_col, pair16, E);
  gemm_kernel<2, 1, 0><<<dim3(E / 128, 128 / 64), 256, 0, stream>>>(
      pair16, W_D0, d0b, y1, nullptr, E, 256, 128, 128);
  gemm_kernel<2, 1, 0><<<dim3(E / 128, 64 / 64), 256, 0, stream>>>(
      y1, W_D1, d1b, y2, nullptr, E, 128, 64, 64);
  final_kernel<<<(E + 7) / 8, 256, 0, stream>>>(y2, d2w, d2b, (float*)d_out, E);

  // tuple tail (edge_index, edge_type, edge_length) if space provided
  tail_kernel<<<(2 * E + 255) / 256, 256, 0, stream>>>(
      eidx, etype, elen, (float*)d_out, E, out_size);
#undef LPAR
}
